// SAE_72378788872670
// MI455X (gfx1250) — matrix operand order repaired and checked by reading, not compiled, not run on hardware
//
#include <hip/hip_runtime.h>
#include <stdint.h>

#define NROWS 8192
#define DIN   768
#define DHID  16384
#define TOPK  32

typedef __attribute__((ext_vector_type(16))) __bf16 v16bf;
typedef __attribute__((ext_vector_type(8)))  __bf16 v8bf;
typedef __attribute__((ext_vector_type(8)))  float  v8f;

// ---------------- 0) fp32 -> bf16 (round-to-nearest-even) ----------------
__global__ __launch_bounds__(256) void k_cvt_bf16(const float* __restrict__ in,
                                                  unsigned short* __restrict__ out,
                                                  int n) {
  int i = blockIdx.x * 256 + threadIdx.x;
  if (i < n) {
    unsigned u = __builtin_bit_cast(unsigned, in[i]);
    unsigned r = (u + 0x7FFFu + ((u >> 16) & 1u)) >> 16;
    out[i] = (unsigned short)r;
  }
}

// ---------------- 1) W_dec [768, 16384] -> W_decT [16384, 768] ----------------
__global__ __launch_bounds__(256) void k_transpose(const float* __restrict__ w,
                                                   float* __restrict__ wt) {
  __shared__ float tile[32][33];
  int c0 = blockIdx.x * 32;   // column block within DHID
  int r0 = blockIdx.y * 32;   // row block within DIN
  int tx = threadIdx.x & 31;
  int ty = threadIdx.x >> 5;  // 0..7
  for (int j = ty; j < 32; j += 8)
    tile[j][tx] = w[(size_t)(r0 + j) * DHID + c0 + tx];
  __syncthreads();
  for (int j = ty; j < 32; j += 8)
    wt[(size_t)(c0 + j) * DIN + r0 + tx] = tile[tx][j];
}

// ---------------- 2) encoder GEMM: latent = relu(x_bf16 . W_enc_bf16^T) -------
// 128x128 block tile per WG (8 waves), K in 64-wide double-buffered LDS stages
// fed by global_load_async_to_lds_b128. Wave (wm,wn) owns a 32x64 sub-tile =
// 2x4 WMMA accumulators. All WMMA operands come from LDS.
__global__ __launch_bounds__(256) void k_enc_gemm(const __bf16* __restrict__ xb,
                                                  const __bf16* __restrict__ wb,
                                                  float* __restrict__ latent) {
  constexpr int BM = 128, BN = 128, BK = 64;
  constexpr int NS = DIN / BK;             // 12 K-stages
  constexpr int STR = BK + 8;              // padded row stride (halfs): 36 dwords
  __shared__ __bf16 shA[2][BM * STR];      // 2 x 18 KB
  __shared__ __bf16 shB[2][BN * STR];      // 2 x 18 KB

  const int tid  = threadIdx.x;
  const int wave = tid >> 5;
  const int lane = tid & 31;
  const int m0 = blockIdx.x * BM;
  const int n0 = blockIdx.y * BN;
  const int wm = wave >> 1;                // 0..3 -> 32-row sub-block
  const int wn = wave & 1;                 // 0..1 -> 64-col sub-block
  const int nl = lane & 15;
  const int hi = lane >> 4;                // wave32 half select (ISA layout)

  // async-stage one 64-wide K panel of A and B into LDS buffer `buf`
  auto issue_stage = [&](int ks, int buf) {
    const __bf16* ga = xb + (size_t)m0 * DIN + ks * BK;
    const __bf16* gbp = wb + (size_t)n0 * DIN + ks * BK;
    #pragma unroll
    for (int i = 0; i < 4; ++i) {              // 4 x 256 = 1024 16B transfers
      int idx = tid + i * 256;
      int r  = idx >> 3;                       // row 0..127
      int c8 = (idx & 7) * 8;                  // half offset 0..56
      unsigned ldsa = (unsigned)(uintptr_t)&shA[buf][r * STR + c8];
      unsigned voff = (unsigned)((r * DIN + c8) * 2);
      asm volatile("global_load_async_to_lds_b128 %0, %1, %2"
                   :: "v"(ldsa), "v"(voff), "s"(ga) : "memory");
      unsigned ldsb = (unsigned)(uintptr_t)&shB[buf][r * STR + c8];
      asm volatile("global_load_async_to_lds_b128 %0, %1, %2"
                   :: "v"(ldsb), "v"(voff), "s"(gbp) : "memory");
    }
  };

  v8f acc[2][4] = {};

  issue_stage(0, 0);
  for (int s = 0; s < NS; ++s) {
    const int buf = s & 1;
    if (s + 1 < NS) {
      issue_stage(s + 1, buf ^ 1);
      asm volatile("s_wait_asynccnt 0x8" ::: "memory");  // stage s done; s+1 in flight
    } else {
      asm volatile("s_wait_asynccnt 0x0" ::: "memory");
    }
    __syncthreads();

    const __bf16* A = &shA[buf][0];
    const __bf16* B = &shB[buf][0];
    #pragma unroll
    for (int kk = 0; kk < 2; ++kk) {           // two 32-wide WMMA k-steps
      const int k32 = kk * 32;
      v16bf af[2];
      #pragma unroll
      for (int mt = 0; mt < 2; ++mt) {
        // A fragment (ISA 16-bit A layout): row = lane%16,
        // elements 0..7 at K = k32 + 8*hi, elements 8..15 at +16.
        const __bf16* ap = A + (wm * 32 + mt * 16 + nl) * STR + k32 + hi * 8;
        v8bf lo  = *(const v8bf*)ap;
        v8bf hi8 = *(const v8bf*)(ap + 16);
        af[mt] = __builtin_shufflevector(lo, hi8,
            0, 1, 2, 3, 4, 5, 6, 7, 8, 9, 10, 11, 12, 13, 14, 15);
      }
      #pragma unroll
      for (int nt = 0; nt < 4; ++nt) {
        // B fragment: col = lane%16, elements 0..7 at K = k32 + 8*hi, elements 8..15 at +16.
        const __bf16* bp = B + (wn * 64 + nt * 16 + nl) * STR + k32 + hi * 8;
        v8bf blo = *(const v8bf*)bp;
        v8bf bhi = *(const v8bf*)(bp + 16);
        v16bf bf_ = __builtin_shufflevector(blo, bhi,
            0, 1, 2, 3, 4, 5, 6, 7, 8, 9, 10, 11, 12, 13, 14, 15);
        acc[0][nt] = __builtin_amdgcn_wmma_f32_16x16x32_bf16(
            false, af[0], false, bf_, (short)0, acc[0][nt], false, false);
        acc[1][nt] = __builtin_amdgcn_wmma_f32_16x16x32_bf16(
            false, af[1], false, bf_, (short)0, acc[1][nt], false, false);
      }
    }
    __syncthreads();   // all waves done reading before buffer is re-filled
  }

  // D layout: VGPR j holds (M = j + 8*hi, N = lane%16). ReLU + store.
  #pragma unroll
  for (int mt = 0; mt < 2; ++mt) {
    #pragma unroll
    for (int j = 0; j < 8; ++j) {
      const int gm = m0 + wm * 32 + mt * 16 + hi * 8 + j;
      float* orow = latent + (size_t)gm * DHID + n0 + wn * 64 + nl;
      #pragma unroll
      for (int nt = 0; nt < 4; ++nt) {
        float v = acc[mt][nt][j];
        orow[nt * 16] = v > 0.f ? v : 0.f;
      }
    }
  }
}

// ---------------- 3) exact top-32 per row (radix select on float bits) --------
__global__ __launch_bounds__(256) void k_topk(float* __restrict__ latent,
                                              float* __restrict__ tkv,
                                              int* __restrict__ tki) {
  __shared__ unsigned srow[DHID];     // 64 KB row copy (bits)
  __shared__ unsigned shist[256];
  __shared__ unsigned sscan[256];
  __shared__ unsigned spfx;
  __shared__ int swant;

  const int tid = threadIdx.x;
  const int row = blockIdx.x;
  float* lrow = latent + (size_t)row * DHID;

  {
    const uint4* src = (const uint4*)lrow;
    uint4* dst = (uint4*)srow;
    for (int i = tid; i < DHID / 4; i += 256) dst[i] = src[i];
  }
  if (tid == 0) { spfx = 0u; swant = TOPK; }
  __syncthreads();

  // Post-ReLU values are >= 0, so IEEE bits order == unsigned order.
  for (int pass = 0; pass < 4; ++pass) {
    const int b = 24 - pass * 8;
    shist[tid] = 0u;
    __syncthreads();
    const unsigned pfx = spfx;
    for (int i = tid; i < DHID; i += 256) {
      unsigned v = srow[i];
      bool match = (pass == 0) || ((v >> (b + 8)) == pfx);
      if (match) atomicAdd(&shist[(v >> b) & 255u], 1u);
    }
    __syncthreads();
    if (tid == 0) {
      int want = swant;
      unsigned acc = 0u;
      int t = 255;
      for (; t > 0; --t) {
        unsigned h = shist[t];
        if (acc + h >= (unsigned)want) break;
        acc += h;
      }
      spfx = (pfx << 8) | (unsigned)t;
      swant = want - (int)acc;
    }
    __syncthreads();
  }
  const unsigned thr = spfx;     // bits of the 32nd-largest value
  const int eqwant = swant;      // #values == thr still to keep

  // Deterministic selection: keep all > thr; among == thr keep lowest indices.
  const int base = tid * (DHID / 256);  // contiguous block of 64 per thread
  unsigned cnt_gt = 0, cnt_eq = 0;
  for (int j = 0; j < (DHID / 256); ++j) {
    unsigned v = srow[base + j];
    cnt_gt += (v > thr);
    cnt_eq += (v == thr);
  }
  sscan[tid] = cnt_eq;
  __syncthreads();
  for (int off = 1; off < 256; off <<= 1) {       // inclusive scan
    unsigned add = (tid >= off) ? sscan[tid - off] : 0u;
    __syncthreads();
    sscan[tid] += add;
    __syncthreads();
  }
  unsigned eq_excl = sscan[tid] - cnt_eq;
  __syncthreads();

  int eq_keep = eqwant - (int)eq_excl;
  if (eq_keep < 0) eq_keep = 0;
  if (eq_keep > (int)cnt_eq) eq_keep = (int)cnt_eq;
  unsigned kept = cnt_gt + (unsigned)eq_keep;

  sscan[tid] = kept;
  __syncthreads();
  for (int off = 1; off < 256; off <<= 1) {
    unsigned add = (tid >= off) ? sscan[tid - off] : 0u;
    __syncthreads();
    sscan[tid] += add;
    __syncthreads();
  }
  unsigned slot = sscan[tid] - kept;   // exclusive prefix = first output slot

  int eq_seen = 0;
  for (int j = 0; j < (DHID / 256); ++j) {
    int i = base + j;
    unsigned v = srow[i];
    bool keep = (v > thr);
    if (v == thr) { keep = (eq_seen < eq_keep); eq_seen++; }
    float fv = __builtin_bit_cast(float, v);
    lrow[i] = keep ? fv : 0.0f;
    if (keep) {
      tkv[row * TOPK + slot] = fv;
      tki[row * TOPK + slot] = i;
      slot++;
    }
  }
}

// ---------------- 4) sparse decode: recon[n,:] = sum_k val*W_decT[idx,:] ------
__global__ __launch_bounds__(256) void k_decode(const float* __restrict__ wdecT,
                                                const float* __restrict__ tkv,
                                                const int* __restrict__ tki,
                                                float* __restrict__ recon) {
  const int row = blockIdx.x;
  const int d = threadIdx.x;
  float a0 = 0.f, a1 = 0.f, a2 = 0.f;
  #pragma unroll 4
  for (int k = 0; k < TOPK; ++k) {
    float v = tkv[row * TOPK + k];     // uniform per block -> scalarized
    int h   = tki[row * TOPK + k];
    const float* wr = wdecT + (size_t)h * DIN;
    a0 += v * wr[d];
    a1 += v * wr[d + 256];
    a2 += v * wr[d + 512];
  }
  float* out = recon + (size_t)row * DIN;
  out[d] = a0; out[d + 256] = a1; out[d + 512] = a2;
}

// ---------------- host glue ----------------
extern "C" void kernel_launch(void* const* d_in, const int* in_sizes, int n_in,
                              void* d_out, int out_size, void* d_ws, size_t ws_size,
                              hipStream_t stream) {
  const float* x    = (const float*)d_in[0];   // [8192, 768]
  const float* Wenc = (const float*)d_in[1];   // [16384, 768]
  const float* Wdec = (const float*)d_in[2];   // [768, 16384]

  float* latent = (float*)d_out;                           // [8192, 16384]
  float* recon  = latent + (size_t)NROWS * DHID;           // [8192, 768]

  char* ws = (char*)d_ws;                                  // ~88 MB used
  unsigned short* xb  = (unsigned short*)(ws);             // x bf16: 12,582,912 B
  unsigned short* wbe = (unsigned short*)(ws + 12582912);  // W_enc bf16: 25,165,824 B
  float* wdecT        = (float*)(ws + 37748736);           // W_decT f32: 50,331,648 B
  float* tkv          = (float*)(ws + 88080384);           // 1,048,576 B
  int*   tki          = (int*)  (ws + 89128960);           // 1,048,576 B

  {
    int n = NROWS * DIN;
    k_cvt_bf16<<<dim3((n + 255) / 256), dim3(256), 0, stream>>>(x, xb, n);
  }
  {
    int n = DHID * DIN;
    k_cvt_bf16<<<dim3((n + 255) / 256), dim3(256), 0, stream>>>(Wenc, wbe, n);
  }
  k_transpose<<<dim3(DHID / 32, DIN / 32), dim3(256), 0, stream>>>(Wdec, wdecT);
  k_enc_gemm<<<dim3(NROWS / 128, DHID / 128), dim3(256), 0, stream>>>(
      (const __bf16*)xb, (const __bf16*)wbe, latent);
  k_topk<<<dim3(NROWS), dim3(256), 0, stream>>>(latent, tkv, tki);
  k_decode<<<dim3(NROWS), dim3(256), 0, stream>>>(wdecT, tkv, tki, recon);
}
